// LearnableKuramotoBank_6906307412147
// MI455X (gfx1250) — matrix operand order repaired and checked by reading, not compiled, not run on hardware
//
#include <hip/hip_runtime.h>
#include <hip/hip_bf16.h>
#include <math.h>

// ---------------------------------------------------------------------------
// LearnableKuramotoBank on MI455X (gfx1250)
//
// Strategy:
//   * One streaming pass over f32 logits: eff_adj = sigmoid(logits) * !diag *
//     precision[j], stored as fp16 (128 MB, fits in 192 MB L2) + per-chunk
//     row sums. Falls back to on-the-fly recompute if workspace too small.
//   * Per RK stage: one WMMA matvec pass computing BOTH eff_adj@cos and
//     eff_adj@sin in the same v_wmma_f32_16x16x32_f16 (cos -> B col 0,
//     sin -> B col 1, cols 2..15 read a zero pad with stride 0 -> the loop
//     body is branch-free: 4 loads + 1 WMMA).
//   * Deterministic: K split into 8 chunks with per-chunk partial outputs,
//     reduced in the O(N) finish kernel (no float atomics).
// ---------------------------------------------------------------------------

#define N_OSC 8192
#define DT_STEP 0.01f
#define KCH 8            // K-chunks per matvec (= waves per block) and build grid.x
#define TILE_ROWS 16

typedef _Float16 half4_t __attribute__((ext_vector_type(4)));
typedef _Float16 half8_t __attribute__((ext_vector_type(8)));
typedef _Float16 v16h    __attribute__((ext_vector_type(16)));
typedef float    v8f     __attribute__((ext_vector_type(8)));

__device__ __forceinline__ float sigmoidf_(float x) {
    return 1.0f / (1.0f + __expf(-x));
}

// ---------------------------------------------------------------------------
// Build pass: sigmoid + diag mask + precision scale -> fp16 matrix (optional)
// and per-chunk row sums (wsump[chunk][row]).  grid = (KCH, N), block = 256,
// 4 columns per thread (1024 columns per block).
// ---------------------------------------------------------------------------
template <bool STORE_HALF>
__global__ __launch_bounds__(256)
void build_adj(const float* __restrict__ logits,
               const float* __restrict__ precision,
               _Float16* __restrict__ Ah,
               float* __restrict__ wsump)
{
    const int row = blockIdx.y;
    const int c0  = (blockIdx.x * 256 + threadIdx.x) * 4;
    const size_t base = (size_t)row * N_OSC + c0;

    const float4 lg = *(const float4*)(logits + base);
    const float4 pr = *(const float4*)(precision + c0);

    float v0 = (row == c0 + 0) ? 0.0f : sigmoidf_(lg.x) * pr.x;
    float v1 = (row == c0 + 1) ? 0.0f : sigmoidf_(lg.y) * pr.y;
    float v2 = (row == c0 + 2) ? 0.0f : sigmoidf_(lg.z) * pr.z;
    float v3 = (row == c0 + 3) ? 0.0f : sigmoidf_(lg.w) * pr.w;

    if constexpr (STORE_HALF) {
        half4_t h;
        h[0] = (_Float16)v0; h[1] = (_Float16)v1;
        h[2] = (_Float16)v2; h[3] = (_Float16)v3;
        *(half4_t*)(Ah + base) = h;
    }

    __shared__ float red[256];
    red[threadIdx.x] = v0 + v1 + v2 + v3;
    __syncthreads();
    for (int s = 128; s > 0; s >>= 1) {
        if (threadIdx.x < (unsigned)s) red[threadIdx.x] += red[threadIdx.x + s];
        __syncthreads();
    }
    if (threadIdx.x == 0)
        wsump[(size_t)blockIdx.x * N_OSC + row] = red[0];
}

// ---------------------------------------------------------------------------
// Stage 0 prep: cos/sin of initial phases, zero RK accumulator and B zero-pad.
// ---------------------------------------------------------------------------
__global__ void stage_init(const float* __restrict__ phases,
                           float* __restrict__ cf, float* __restrict__ sf,
                           _Float16* __restrict__ ch, _Float16* __restrict__ sh,
                           float* __restrict__ ksum,
                           _Float16* __restrict__ zbuf)
{
    int i = blockIdx.x * blockDim.x + threadIdx.x;
    if (i >= N_OSC) return;
    if (i < 64) zbuf[i] = (_Float16)0.0f;   // zero pad for B cols 2..15
    float p = phases[i];
    float c = cosf(p), s = sinf(p);
    cf[i] = c; sf[i] = s;
    ch[i] = (_Float16)c; sh[i] = (_Float16)s;
    ksum[i] = 0.0f;
}

// ---------------------------------------------------------------------------
// WMMA matvec: each wave -> 16 output rows x 1024-wide K chunk.
//   A fragment (16-bit 16x32, ISA 7.12.2): lane row = lane&15,
//     halfs[0..7]  = A[row][kb + hi*8 .. +7]
//     halfs[8..15] = A[row][kb + hi*8 + 16 .. +23]        (hi = lane>>4)
//   B fragment (col per lane, ISA 7.12.4 pattern): lane col = lane&15,
//     halfs[0..7] = B[kb + hi*8 .. +7][col], halfs[8..15] = B[kb + hi*8 + 16 .. +23][col];  col0 = cos, col1 = sin,
//     cols 2..15 read a 32-half zero pad with per-lane stride 0 (branch-free).
//   D (f32 16x16): vgpr r, col = lane&15, row = hi*8 + r.
// Partials stored per chunk -> no atomics, deterministic.
// ---------------------------------------------------------------------------
template <bool FROM_HALF>
__global__ __launch_bounds__(256)
void matvec_wmma(const _Float16* __restrict__ Ah,
                 const float* __restrict__ logits,
                 const float* __restrict__ precision,
                 const _Float16* __restrict__ ch,
                 const _Float16* __restrict__ sh,
                 const _Float16* __restrict__ zbuf,
                 float* __restrict__ Acp, float* __restrict__ Asp)
{
    const int lane    = threadIdx.x & 31;
    const int wave    = threadIdx.x >> 5;          // K chunk id, 0..KCH-1
    const int rowbase = blockIdx.x * TILE_ROWS;
    const int col     = lane & 15;
    const int hi      = lane >> 4;
    const int arow    = rowbase + col;             // A row held by this lane
    const int aoff    = hi * 8;
    const int kbeg    = wave * (N_OSC / KCH);
    const int kend    = kbeg + (N_OSC / KCH);

    // Branch-free B source: col 0 -> cos, col 1 -> sin, cols 2..15 -> zero pad
    const _Float16* bsrc;
    int binc;
    if (col == 0)      { bsrc = ch + kbeg + hi * 8; binc = 32; }
    else if (col == 1) { bsrc = sh + kbeg + hi * 8; binc = 32; }
    else               { bsrc = zbuf;                binc = 0;  }

    v8f acc = {};
    for (int kb = kbeg; kb < kend; kb += 32) {
        union { v16h v; half8_t h8[2]; _Float16 h[16]; } a;
        if constexpr (FROM_HALF) {
            const _Float16* ap = Ah + (size_t)arow * N_OSC + kb + aoff;
            a.h8[0] = *(const half8_t*)(ap);
            a.h8[1] = *(const half8_t*)(ap + 16);
        } else {
            const float* lp = logits + (size_t)arow * N_OSC + kb + aoff;
            float t[16];
            *(float4*)(t + 0)  = *(const float4*)(lp + 0);
            *(float4*)(t + 4)  = *(const float4*)(lp + 4);
            *(float4*)(t + 8)  = *(const float4*)(lp + 16);
            *(float4*)(t + 12) = *(const float4*)(lp + 20);
#pragma unroll
            for (int u = 0; u < 16; ++u) {
                int j = kb + aoff + (u < 8 ? u : u + 8);
                float v = (arow == j) ? 0.0f : sigmoidf_(t[u]) * precision[j];
                a.h[u] = (_Float16)v;
            }
        }

        union { v16h v; half8_t h8[2]; } bu;
        bu.h8[0] = *(const half8_t*)(bsrc);
        bu.h8[1] = *(const half8_t*)(bsrc + 16);
        const v16h b = bu.v;
        bsrc += binc;

        // D = A(16x32 fp16) * B(32x16 fp16) + C(16x16 f32)
        acc = __builtin_amdgcn_wmma_f32_16x16x32_f16(
            /*neg_a=*/false, a.v, /*neg_b=*/false, b,
            /*c_mod=*/(short)0, acc, /*reuse_a=*/false, /*reuse_b=*/false);
    }

    if (col < 2) {
        float* dst = (col == 0) ? Acp : Asp;
        const size_t off = (size_t)wave * N_OSC + rowbase + hi * 8;
#pragma unroll
        for (int r = 0; r < 8; ++r) dst[off + r] = acc[r];
    }
}

// ---------------------------------------------------------------------------
// Per-stage epilogue: reduce chunk partials, form k_m, accumulate RK sum,
// prepare next stage's cos/sin (or emit final phases at stage 3).
// ---------------------------------------------------------------------------
__global__ void stage_finish(int stage,
                             const float* __restrict__ phases,
                             const float* __restrict__ omega,
                             const float* __restrict__ ext,
                             const float* __restrict__ Kp,
                             const float* __restrict__ wsump,
                             const float* __restrict__ Acp,
                             const float* __restrict__ Asp,
                             float* __restrict__ cf, float* __restrict__ sf,
                             _Float16* __restrict__ ch, _Float16* __restrict__ sh,
                             float* __restrict__ ksum,
                             float* __restrict__ out)
{
    int i = blockIdx.x * blockDim.x + threadIdx.x;
    if (i >= N_OSC) return;

    float ac = 0.0f, as = 0.0f, w = 0.0f;
#pragma unroll
    for (int c = 0; c < KCH; ++c) {
        ac += Acp[(size_t)c * N_OSC + i];
        as += Asp[(size_t)c * N_OSC + i];
        w  += wsump[(size_t)c * N_OSC + i];
    }
    w = fmaxf(w, 1e-8f);

    const float Kc = Kp[0];
    const float k  = omega[i] + Kc * ((sf[i] * ac - cf[i] * as) / w) + ext[i];
    const float wk = (stage == 0 || stage == 3) ? 1.0f : 2.0f;
    const float ks = ksum[i] + wk * k;

    if (stage == 3) {
        out[i] = phases[i] + (DT_STEP / 6.0f) * ks;
    } else {
        ksum[i] = ks;
        const float cn = (stage == 2) ? 1.0f : 0.5f;
        const float pn = phases[i] + cn * DT_STEP * k;
        const float c = cosf(pn), s = sinf(pn);
        cf[i] = c; sf[i] = s;
        ch[i] = (_Float16)c; sh[i] = (_Float16)s;
    }
}

// ---------------------------------------------------------------------------
// Host launcher
// Inputs: 0 external_input[N], 1 phases[N], 2 natural_frequencies[N],
//         3 coupling_strength[1], 4 adjacency_logits[N*N], 5 precision[N]
// ---------------------------------------------------------------------------
extern "C" void kernel_launch(void* const* d_in, const int* in_sizes, int n_in,
                              void* d_out, int out_size, void* d_ws, size_t ws_size,
                              hipStream_t stream)
{
    const float* ext    = (const float*)d_in[0];
    const float* phases = (const float*)d_in[1];
    const float* omega  = (const float*)d_in[2];
    const float* Kp     = (const float*)d_in[3];
    const float* logits = (const float*)d_in[4];
    const float* prec   = (const float*)d_in[5];
    float* out = (float*)d_out;

    const size_t n = N_OSC;
    uint8_t* w = (uint8_t*)d_ws;
    float* wsump = (float*)w;  w += KCH * n * sizeof(float);
    float* Acp   = (float*)w;  w += KCH * n * sizeof(float);
    float* Asp   = (float*)w;  w += KCH * n * sizeof(float);
    float* cf    = (float*)w;  w += n * sizeof(float);
    float* sf    = (float*)w;  w += n * sizeof(float);
    float* ksum  = (float*)w;  w += n * sizeof(float);
    _Float16* ch   = (_Float16*)w;  w += n * sizeof(_Float16);
    _Float16* sh   = (_Float16*)w;  w += n * sizeof(_Float16);
    _Float16* zbuf = (_Float16*)w;  w += 64 * sizeof(_Float16);

    size_t off = (size_t)(w - (uint8_t*)d_ws);
    off = (off + 255) & ~(size_t)255;
    _Float16* Ah = (_Float16*)((uint8_t*)d_ws + off);
    const bool cacheA = ws_size >= off + n * n * sizeof(_Float16);

    stage_init<<<dim3((unsigned)(n / 256)), 256, 0, stream>>>(phases, cf, sf, ch, sh, ksum, zbuf);

    if (cacheA)
        build_adj<true ><<<dim3(KCH, (unsigned)n), 256, 0, stream>>>(logits, prec, Ah, wsump);
    else
        build_adj<false><<<dim3(KCH, (unsigned)n), 256, 0, stream>>>(logits, prec, nullptr, wsump);

    for (int stage = 0; stage < 4; ++stage) {
        if (cacheA)
            matvec_wmma<true ><<<dim3((unsigned)(n / TILE_ROWS)), 256, 0, stream>>>(
                Ah, logits, prec, ch, sh, zbuf, Acp, Asp);
        else
            matvec_wmma<false><<<dim3((unsigned)(n / TILE_ROWS)), 256, 0, stream>>>(
                nullptr, logits, prec, ch, sh, zbuf, Acp, Asp);

        stage_finish<<<dim3((unsigned)(n / 256)), 256, 0, stream>>>(
            stage, phases, omega, ext, Kp, wsump, Acp, Asp,
            cf, sf, ch, sh, ksum, out);
    }
}
